// MOELayer_6571299962933
// MI455X (gfx1250) — matrix operand order repaired and checked by reading, compiled, not run on hardware
//
#include <hip/hip_runtime.h>
#include <hip/hip_bf16.h>
#include <math.h>

typedef __attribute__((ext_vector_type(16))) _Float16 v16h;
typedef __attribute__((ext_vector_type(8)))  _Float16 v8h;
typedef __attribute__((ext_vector_type(8)))  float    v8f;
typedef int moe_v4i __attribute__((vector_size(16)));   // matches builtin's V4i param

#if __has_builtin(__builtin_amdgcn_global_load_async_to_lds_b128)
#define MOE_USE_ASYNC 1
#else
#define MOE_USE_ASYNC 0
#endif

#define GAS1(p) ((__attribute__((address_space(1))) moe_v4i*)(p))
#define LAS3(p) ((__attribute__((address_space(3))) moe_v4i*)(p))

namespace {

constexpr int E    = 8;     // experts
constexpr int TOPK = 2;
constexpr int D    = 1024;  // model dim
constexpr int H    = 4096;  // hidden per expert

// GEMM tiling: workgroup computes 128(M) x 128(N), K-step 32, 8 waves of 32x64.
constexpr int MT  = 128;
constexpr int NT  = 128;
constexpr int KT  = 32;
constexpr int LDT = 40;     // padded LDS stride in f16 -> conflict-free b128 frag reads

__device__ inline void wait_async0() {
#if __has_builtin(__builtin_amdgcn_s_wait_asynccnt)
  __builtin_amdgcn_s_wait_asynccnt(0);
#else
  asm volatile("s_wait_asynccnt 0x0" ::: "memory");
#endif
}

__device__ inline v16h cat8(v8h lo, v8h hi) {
  return __builtin_shufflevector(lo, hi, 0,1,2,3,4,5,6,7,8,9,10,11,12,13,14,15);
}

// A fragment (16x32 f16, MxK): lane L holds row m=L%16;
// VGPRs 0-3 -> k = 8*(L/16)+0..7 ; VGPRs 4-7 -> k = 16+8*(L/16)+0..7
__device__ inline v16h ld_frag_a(const _Float16* t, int row, int half) {
  const int k = half * 8;
  v8h lo = *(const v8h*)(t + row * LDT + k);
  v8h hi = *(const v8h*)(t + row * LDT + 16 + k);
  return cat8(lo, hi);
}

// B fragment (32x16 f16, KxN), tile stored [n][k]: lane L holds col n=L%16;
// VGPRs 0-3 -> k = 8*(L/16)+0..7 ; VGPRs 4-7 -> k = 16+8*(L/16)+0..7
__device__ inline v16h ld_frag_b(const _Float16* t, int row, int half) {
  const int k = half * 8;
  v8h lo = *(const v8h*)(t + row * LDT + k);
  v8h hi = *(const v8h*)(t + row * LDT + k + 16);
  return cat8(lo, hi);
}

#define WMMA_F16(A, B, C) \
  __builtin_amdgcn_wmma_f32_16x16x32_f16(false, (A), false, (B), (short)0, (C), false, false)

// Preload all fragments, then 8 back-to-back WMMAs (pipelines in the XDL).
__device__ inline void mma_step(const _Float16* As, const _Float16* Bs,
                                int wm, int wn, int fr, int half, v8f (&acc)[2][4]) {
  v16h a0 = ld_frag_a(As, wm * 32 + fr,      half);
  v16h a1 = ld_frag_a(As, wm * 32 + 16 + fr, half);
  v16h b0 = ld_frag_b(Bs, wn * 64 +  0 + fr, half);
  v16h b1 = ld_frag_b(Bs, wn * 64 + 16 + fr, half);
  v16h b2 = ld_frag_b(Bs, wn * 64 + 32 + fr, half);
  v16h b3 = ld_frag_b(Bs, wn * 64 + 48 + fr, half);
  acc[0][0] = WMMA_F16(a0, b0, acc[0][0]);
  acc[1][0] = WMMA_F16(a1, b0, acc[1][0]);
  acc[0][1] = WMMA_F16(a0, b1, acc[0][1]);
  acc[1][1] = WMMA_F16(a1, b1, acc[1][1]);
  acc[0][2] = WMMA_F16(a0, b2, acc[0][2]);
  acc[1][2] = WMMA_F16(a1, b2, acc[1][2]);
  acc[0][3] = WMMA_F16(a0, b3, acc[0][3]);
  acc[1][3] = WMMA_F16(a1, b3, acc[1][3]);
}

// ---------------- gating: logits -> softmax -> top2 -> normalized gates ----
__global__ __launch_bounds__(256) void moe_gate(const float* __restrict__ x,
                                                const float* __restrict__ gate_w,
                                                int* __restrict__ topk_e,
                                                float* __restrict__ topk_g, int N) {
  __shared__ float gw[E * D];
  for (int i = threadIdx.x; i < E * D; i += 256) gw[i] = gate_w[i];
  __syncthreads();
  const int t = blockIdx.x * 256 + threadIdx.x;
  if (t >= N) return;
  const float4* xr = (const float4*)(x + (size_t)t * D);
  float lg[E];
#pragma unroll
  for (int e = 0; e < E; ++e) lg[e] = 0.f;
  for (int q = 0; q < D / 4; ++q) {
    float4 xv = xr[q];
#pragma unroll
    for (int e = 0; e < E; ++e) {
      const float* g = &gw[e * D + q * 4];
      lg[e] += xv.x * g[0] + xv.y * g[1] + xv.z * g[2] + xv.w * g[3];
    }
  }
  float m = lg[0];
#pragma unroll
  for (int e = 1; e < E; ++e) m = fmaxf(m, lg[e]);
  float p[E];
#pragma unroll
  for (int e = 0; e < E; ++e) p[e] = __expf(lg[e] - m);
  int e0 = 0; float v0 = p[0];
#pragma unroll
  for (int e = 1; e < E; ++e) if (p[e] > v0) { v0 = p[e]; e0 = e; }
  int e1 = (e0 == 0) ? 1 : 0; float v1 = p[e1];
#pragma unroll
  for (int e = 0; e < E; ++e) if (e != e0 && p[e] > v1) { v1 = p[e]; e1 = e; }
  const float inv = 1.f / (v0 + v1);
  topk_e[t * TOPK + 0] = e0; topk_e[t * TOPK + 1] = e1;
  topk_g[t * TOPK + 0] = v0 * inv; topk_g[t * TOPK + 1] = v1 * inv;
}

// -------- deterministic slot-major rank (matches reference cumsum order) ----
__global__ __launch_bounds__(1024) void moe_assign(const int* __restrict__ topk_e,
                                                   const float* __restrict__ topk_g,
                                                   int* __restrict__ slot2tok,
                                                   int* __restrict__ comb_idx,
                                                   float* __restrict__ comb_w,
                                                   int N, int cap) {
  __shared__ int gbase[E];
  __shared__ int waveCnt[32][E];
  __shared__ int waveBase[32][E];
  __shared__ int chunkTot[E];
  const int tid = threadIdx.x, lane = tid & 31, wv = tid >> 5;
  for (int s = tid; s < E * cap; s += 1024) slot2tok[s] = -1;
  if (tid < E) gbase[tid] = 0;
  __syncthreads();
  const int total = N * TOPK;
  for (int a0 = 0; a0 < total; a0 += 1024) {
    const int a = a0 + tid;
    const bool act = a < total;
    const int j = act ? a / N : 0;
    const int i = act ? a - j * N : 0;
    const int e = act ? topk_e[i * TOPK + j] : -1;
    unsigned mymask = 0;
#pragma unroll
    for (int ee = 0; ee < E; ++ee) {
      unsigned msk = (unsigned)__builtin_amdgcn_ballot_w32(e == ee);
      if (ee == e) mymask = msk;
      if (lane == ee) waveCnt[wv][ee] = __builtin_popcount(msk);
    }
    __syncthreads();
    if (tid < E) {
      int s = 0;
      for (int w = 0; w < 32; ++w) { waveBase[w][tid] = s; s += waveCnt[w][tid]; }
      chunkTot[tid] = s;
    }
    __syncthreads();
    if (act) {
      const int before = __builtin_popcount(mymask & ((1u << lane) - 1u));
      const int loc = gbase[e] + waveBase[wv][e] + before;
      const bool valid = loc < cap;
      const int flat = e * cap + loc;
      if (valid) slot2tok[flat] = i;
      comb_idx[i * TOPK + j] = valid ? flat : 0;
      comb_w[i * TOPK + j]   = valid ? topk_g[i * TOPK + j] : 0.f;
    }
    __syncthreads();
    if (tid < E) gbase[tid] += chunkTot[tid];
    __syncthreads();
  }
}

// ---------------- FFN1: hid = relu(dispatch(x) @ W1^T + b1), f16 out --------
// Double-buffered LDS + register prefetch: one barrier per K-step, global-load
// latency hidden under the 8 WMMAs of the previous step.
__global__ __launch_bounds__(256) void moe_ffn1(const float* __restrict__ x,
                                                const float* __restrict__ fc1_w,
                                                const float* __restrict__ fc1_b,
                                                const int* __restrict__ slot2tok,
                                                _Float16* __restrict__ hid, int cap) {
  __shared__ alignas(16) _Float16 As[2][MT * LDT];
  __shared__ alignas(16) _Float16 Bs[2][NT * LDT];
  const int e = blockIdx.z, mt = blockIdx.y, nt = blockIdx.x;
  const int tid = threadIdx.x, lane = tid & 31, wv = tid >> 5;
  const int wm = wv & 3, wn = wv >> 2;
  const int half = lane >> 4, fr = lane & 15;

  v8f acc[2][4] = {};

  // staging: 2 threads per row, 16 f32->f16 each
  const int ar = tid >> 1, ac = (tid & 1) * 16;
  const int tok = slot2tok[e * cap + mt * MT + ar];
  const float4* arow = (tok >= 0) ? (const float4*)(x + (size_t)tok * D + ac) : nullptr;
  const float4* brow = (const float4*)(fc1_w + ((size_t)e * H + nt * NT + ar) * D + ac);

  float4 a4[4], b4[4];
  auto ldglob = [&](int s) {
    const int q = s * (KT / 4);
    if (arow) { a4[0] = arow[q]; a4[1] = arow[q + 1]; a4[2] = arow[q + 2]; a4[3] = arow[q + 3]; }
    else { a4[0] = a4[1] = a4[2] = a4[3] = float4{0.f, 0.f, 0.f, 0.f}; }
    b4[0] = brow[q]; b4[1] = brow[q + 1]; b4[2] = brow[q + 2]; b4[3] = brow[q + 3];
  };
  auto stage = [&](int buf) {
    _Float16* ad = &As[buf][ar * LDT + ac];
    _Float16* bd = &Bs[buf][ar * LDT + ac];
#pragma unroll
    for (int jj = 0; jj < 4; ++jj) {
      ad[jj * 4 + 0] = (_Float16)a4[jj].x; ad[jj * 4 + 1] = (_Float16)a4[jj].y;
      ad[jj * 4 + 2] = (_Float16)a4[jj].z; ad[jj * 4 + 3] = (_Float16)a4[jj].w;
      bd[jj * 4 + 0] = (_Float16)b4[jj].x; bd[jj * 4 + 1] = (_Float16)b4[jj].y;
      bd[jj * 4 + 2] = (_Float16)b4[jj].z; bd[jj * 4 + 3] = (_Float16)b4[jj].w;
    }
  };

  constexpr int S = D / KT;
  ldglob(0);
  stage(0);
  ldglob(1);
  __syncthreads();
  int cur = 0;
  for (int s = 0; s < S; ++s, cur ^= 1) {
    if (s + 1 < S) {
      stage(cur ^ 1);
      if (s + 2 < S) ldglob(s + 2);
    }
    mma_step(As[cur], Bs[cur], wm, wn, fr, half, acc);
    __syncthreads();
  }

  const int rbase = mt * MT + wm * 32 + (half << 3);
#pragma unroll
  for (int ni = 0; ni < 4; ++ni) {
    const int hcol = nt * NT + wn * 64 + ni * 16 + fr;
    const float bias = fc1_b[(size_t)e * H + hcol];
#pragma unroll
    for (int mi = 0; mi < 2; ++mi)
#pragma unroll
      for (int v = 0; v < 8; ++v) {
        float val = acc[mi][ni][v] + bias;
        val = val > 0.f ? val : 0.f;
        hid[((size_t)e * cap + rbase + mi * 16 + v) * H + hcol] = (_Float16)val;
      }
  }
}

// ---------------- FFN2: eout = hid @ W2 + b2, f32 out ----------------------
// A-tile is a raw f16 copy -> CDNA5 async global->LDS (ASYNCcnt) when available;
// B-tile (fp32 W2, transposed into [n][k]) uses register prefetch + cvt.
__global__ __launch_bounds__(256) void moe_ffn2(const _Float16* __restrict__ hid,
                                                const float* __restrict__ fc2_w,
                                                const float* __restrict__ fc2_b,
                                                float* __restrict__ eout, int cap) {
  __shared__ alignas(16) _Float16 As[2][MT * LDT];
  __shared__ alignas(16) _Float16 Bs[2][NT * LDT];
  const int e = blockIdx.z, mt = blockIdx.y, nt = blockIdx.x;
  const int tid = threadIdx.x, lane = tid & 31, wv = tid >> 5;
  const int wm = wv & 3, wn = wv >> 2;
  const int half = lane >> 4, fr = lane & 15;

  v8f acc[2][4] = {};

  const int ar = tid >> 1, ac = (tid & 1) * 16;
  const _Float16* asrc = hid + ((size_t)e * cap + mt * MT + ar) * H + ac;
  const int kr = tid >> 3, nc = (tid & 7) * 16;
  const float* bbase = fc2_w + ((size_t)e * H + kr) * D + nt * NT + nc;

  float4 b4[4];
  auto ldB = [&](int s) {
    const float4* p = (const float4*)(bbase + (size_t)s * KT * D);
    b4[0] = p[0]; b4[1] = p[1]; b4[2] = p[2]; b4[3] = p[3];
  };
  auto stageB = [&](int buf) {
#pragma unroll
    for (int jj = 0; jj < 4; ++jj) {
      Bs[buf][(nc + jj * 4 + 0) * LDT + kr] = (_Float16)b4[jj].x;
      Bs[buf][(nc + jj * 4 + 1) * LDT + kr] = (_Float16)b4[jj].y;
      Bs[buf][(nc + jj * 4 + 2) * LDT + kr] = (_Float16)b4[jj].z;
      Bs[buf][(nc + jj * 4 + 3) * LDT + kr] = (_Float16)b4[jj].w;
    }
  };
  auto stageA = [&](int s, int buf) {
    const _Float16* g = asrc + s * KT;
    _Float16* l = &As[buf][ar * LDT + ac];
#if MOE_USE_ASYNC
    __builtin_amdgcn_global_load_async_to_lds_b128(GAS1(g), LAS3(l), 0, 0);
    __builtin_amdgcn_global_load_async_to_lds_b128(GAS1(g), LAS3(l), 16, 0);
#else
    *(v8h*)l       = *(const v8h*)g;
    *(v8h*)(l + 8) = *(const v8h*)(g + 8);
#endif
  };

  constexpr int S = H / KT;
  ldB(0);
  stageB(0);
  stageA(0, 0);
  ldB(1);
#if MOE_USE_ASYNC
  wait_async0();
#endif
  __syncthreads();
  int cur = 0;
  for (int s = 0; s < S; ++s, cur ^= 1) {
    if (s + 1 < S) {
      stageB(cur ^ 1);
      stageA(s + 1, cur ^ 1);
      if (s + 2 < S) ldB(s + 2);
    }
    mma_step(As[cur], Bs[cur], wm, wn, fr, half, acc);
#if MOE_USE_ASYNC
    wait_async0();
#endif
    __syncthreads();
  }

  const int rbase = mt * MT + wm * 32 + (half << 3);
#pragma unroll
  for (int ni = 0; ni < 4; ++ni) {
    const int dcol = nt * NT + wn * 64 + ni * 16 + fr;
    const float bias = fc2_b[(size_t)e * D + dcol];
#pragma unroll
    for (int mi = 0; mi < 2; ++mi)
#pragma unroll
      for (int v = 0; v < 8; ++v)
        eout[((size_t)e * cap + rbase + mi * 16 + v) * D + dcol] = acc[mi][ni][v] + bias;
  }
}

// ---------------- combine: out = g0*eout[slot0] + g1*eout[slot1] -----------
__global__ __launch_bounds__(256) void moe_combine(const float* __restrict__ eout,
                                                   const int* __restrict__ comb_idx,
                                                   const float* __restrict__ comb_w,
                                                   float* __restrict__ out) {
  const int t = blockIdx.x;
  const int i0 = comb_idx[t * TOPK], i1 = comb_idx[t * TOPK + 1];
  const float w0 = comb_w[t * TOPK], w1 = comb_w[t * TOPK + 1];
  const float4* r0 = (const float4*)(eout + (size_t)i0 * D);
  const float4* r1 = (const float4*)(eout + (size_t)i1 * D);
  float4* o = (float4*)(out + (size_t)t * D);
  const int q = threadIdx.x;            // D/4 == 256
  float4 a = r0[q], b = r1[q], r;
  r.x = w0 * a.x + w1 * b.x; r.y = w0 * a.y + w1 * b.y;
  r.z = w0 * a.z + w1 * b.z; r.w = w0 * a.w + w1 * b.w;
  o[q] = r;
}

}  // namespace

extern "C" void kernel_launch(void* const* d_in, const int* in_sizes, int n_in,
                              void* d_out, int out_size, void* d_ws, size_t ws_size,
                              hipStream_t stream) {
  const float* x      = (const float*)d_in[0];
  const float* gate_w = (const float*)d_in[1];
  const float* fc1_w  = (const float*)d_in[2];
  const float* fc1_b  = (const float*)d_in[3];
  const float* fc2_w  = (const float*)d_in[4];
  const float* fc2_b  = (const float*)d_in[5];
  float* out = (float*)d_out;

  const int N   = in_sizes[0] / D;                 // 8192 tokens
  const int cap = TOPK * ((N + E - 1) / E);        // 2048

  char* ws = (char*)d_ws;
  auto carve = [&](size_t bytes) {
    char* p = ws;
    ws += (bytes + 255) & ~(size_t)255;
    return p;
  };
  int*      topk_e = (int*)     carve((size_t)N * TOPK * sizeof(int));
  float*    topk_g = (float*)   carve((size_t)N * TOPK * sizeof(float));
  int*      comb_i = (int*)     carve((size_t)N * TOPK * sizeof(int));
  float*    comb_w = (float*)   carve((size_t)N * TOPK * sizeof(float));
  int*      s2t    = (int*)     carve((size_t)E * cap * sizeof(int));
  _Float16* hid    = (_Float16*)carve((size_t)E * cap * H * sizeof(_Float16));
  float*    eout   = (float*)   carve((size_t)E * cap * D * sizeof(float));

  moe_gate<<<(N + 255) / 256, 256, 0, stream>>>(x, gate_w, topk_e, topk_g, N);
  moe_assign<<<1, 1024, 0, stream>>>(topk_e, topk_g, s2t, comb_i, comb_w, N, cap);
  moe_ffn1<<<dim3(H / NT, cap / MT, E), 256, 0, stream>>>(x, fc1_w, fc1_b, s2t, hid, cap);
  moe_ffn2<<<dim3(D / NT, cap / MT, E), 256, 0, stream>>>(hid, fc2_w, fc2_b, eout, cap);
  moe_combine<<<N, 256, 0, stream>>>(eout, comb_i, comb_w, out);
}
